// SimpleRNN_58523224375383
// MI455X (gfx1250) — matrix operand order repaired and checked by reading, not compiled, not run on hardware
//
#include <hip/hip_runtime.h>
#include <hip/hip_bf16.h>
#include <cstdint>

typedef _Float16 v16h __attribute__((ext_vector_type(16)));
typedef _Float16 v8h  __attribute__((ext_vector_type(8)));
typedef float    v8f  __attribute__((ext_vector_type(8)));

#define BB 64
#define TT 512
#define II 512
#define HH 1024
#define OO 512

// Scheduling fence: nothing may be reordered across this point. Used to pin
// the prefetch->compute pipeline stages so the scheduler cannot sink loads
// next to their consuming WMMA (which collapses the double buffer and exposes
// full load latency via s_wait_loadcnt 0x0).
#define SCHED_FENCE() __builtin_amdgcn_sched_barrier(0)

// ---------------------------------------------------------------------------
// Fragment layout for V_WMMA_F32_16X16X32_F16 (wave32), per ISA 7.12.2.
// A: 16x32 (MxK) f16 — lane m%16 holds row m; lanes 16-31 take K 16-31 (contiguous, k = 16*hi + s).
// B: 32x16 (KxN) f16 — lane n holds COLUMN n = one row of row-major W[N,K].
// ---------------------------------------------------------------------------
__device__ __forceinline__ v16h load_afrag_p(const _Float16* __restrict__ pa, int k) {
    union { v16h v; v8h h[2]; } u;
    u.h[0] = *(const v8h*)(pa + k);        // K = k+16*hi .. +7
    u.h[1] = *(const v8h*)(pa + k + 8);    // K = k+16*hi+8 .. +7
    return u.v;
}

__device__ __forceinline__ v16h load_bfrag_p(const _Float16* __restrict__ pb, int k) {
    return *(const v16h*)(pb + k);         // 32 contiguous bytes
}

// ---------------------------------------------------------------------------
// 16xK @ Kx64 strip, manual unroll-by-2 ping-pong double buffering with
// sched barriers delimiting the pipeline stages:
//   [prefetch other buf: 10 loads] || [4 WMMAs on current buf]
// First use of a buffer is 10 loads after its issue -> partial waits only.
// Requires K % 64 == 0 (true for 128 / 512 / 1024).
// ---------------------------------------------------------------------------
template <int K>
__device__ __forceinline__ void gemm_strip(const _Float16* __restrict__ A, int lda, int m0,
                                           const _Float16* __restrict__ W, int ldb, int n0,
                                           int lane, v8f acc[4]) {
    const int r  = lane & 15;
    const int hi = lane >> 4;
    const _Float16* pa  = A + (size_t)(m0 + r) * lda + hi * 16;
    const _Float16* pb0 = W + (size_t)(n0 +  0 + r) * ldb + hi * 16;
    const _Float16* pb1 = W + (size_t)(n0 + 16 + r) * ldb + hi * 16;
    const _Float16* pb2 = W + (size_t)(n0 + 32 + r) * ldb + hi * 16;
    const _Float16* pb3 = W + (size_t)(n0 + 48 + r) * ldb + hi * 16;

    v16h a[2];
    v16h b[2][4];
    a[0]    = load_afrag_p(pa, 0);
    b[0][0] = load_bfrag_p(pb0, 0);
    b[0][1] = load_bfrag_p(pb1, 0);
    b[0][2] = load_bfrag_p(pb2, 0);
    b[0][3] = load_bfrag_p(pb3, 0);
    SCHED_FENCE();

    for (int it = 0; it < K / 64 - 1; ++it) {
        // stage: prefetch buf1 @ +32
        a[1]    = load_afrag_p(pa, 32);
        b[1][0] = load_bfrag_p(pb0, 32);
        b[1][1] = load_bfrag_p(pb1, 32);
        b[1][2] = load_bfrag_p(pb2, 32);
        b[1][3] = load_bfrag_p(pb3, 32);
        SCHED_FENCE();
        // stage: consume buf0
#pragma unroll
        for (int nt = 0; nt < 4; ++nt)
            acc[nt] = __builtin_amdgcn_wmma_f32_16x16x32_f16(
                false, a[0], false, b[0][nt], (short)0, acc[nt], false, false);
        SCHED_FENCE();
        // stage: prefetch buf0 @ +64
        a[0]    = load_afrag_p(pa, 64);
        b[0][0] = load_bfrag_p(pb0, 64);
        b[0][1] = load_bfrag_p(pb1, 64);
        b[0][2] = load_bfrag_p(pb2, 64);
        b[0][3] = load_bfrag_p(pb3, 64);
        SCHED_FENCE();
        // stage: consume buf1
#pragma unroll
        for (int nt = 0; nt < 4; ++nt)
            acc[nt] = __builtin_amdgcn_wmma_f32_16x16x32_f16(
                false, a[1], false, b[1][nt], (short)0, acc[nt], false, false);
        SCHED_FENCE();
        pa += 64; pb0 += 64; pb1 += 64; pb2 += 64; pb3 += 64;
    }
    // peeled tail: last 64 K-values
    a[1]    = load_afrag_p(pa, 32);
    b[1][0] = load_bfrag_p(pb0, 32);
    b[1][1] = load_bfrag_p(pb1, 32);
    b[1][2] = load_bfrag_p(pb2, 32);
    b[1][3] = load_bfrag_p(pb3, 32);
    SCHED_FENCE();
#pragma unroll
    for (int nt = 0; nt < 4; ++nt)
        acc[nt] = __builtin_amdgcn_wmma_f32_16x16x32_f16(
            false, a[0], false, b[0][nt], (short)0, acc[nt], false, false);
#pragma unroll
    for (int nt = 0; nt < 4; ++nt)
        acc[nt] = __builtin_amdgcn_wmma_f32_16x16x32_f16(
            false, a[1], false, b[1][nt], (short)0, acc[nt], false, false);
}

// ---------------------------------------------------------------------------
// fp32 -> f16 conversion (grid-stride)
// ---------------------------------------------------------------------------
__global__ __launch_bounds__(256) void f32_to_f16_kernel(const float* __restrict__ s,
                                                         _Float16* __restrict__ d, int n) {
    int i = blockIdx.x * blockDim.x + threadIdx.x;
    int stride = gridDim.x * blockDim.x;
    for (; i < n; i += stride) d[i] = (_Float16)s[i];
}

// ---------------------------------------------------------------------------
// Phase 1: xh[t,b,:] = x[b,t,:] @ Wx^T + bx + bh      (M=B*T, K=I, N=H)
// grid: (H/64, M/128), block 256 (8 waves, each: 16 rows x 64 cols)
// ---------------------------------------------------------------------------
__global__ __launch_bounds__(256, 1) __attribute__((amdgpu_waves_per_eu(1)))
void xh_gemm_kernel(const _Float16* __restrict__ Xf,
                    const _Float16* __restrict__ Wxf,
                    const float* __restrict__ bx,
                    const float* __restrict__ bh,
                    float* __restrict__ xh) {
    int lane = threadIdx.x & 31;
    int wave = threadIdx.x >> 5;
    int m0 = (blockIdx.y * 8 + wave) * 16;
    int n0 = blockIdx.x * 64;

    v8f acc[4] = {};
    gemm_strip<II>(Xf, II, m0, Wxf, II, n0, lane, acc);

    int hi = lane >> 4, c = lane & 15;
#pragma unroll
    for (int nt = 0; nt < 4; ++nt) {
        int col = n0 + nt * 16 + c;
        float bias = bx[col] + bh[col];
#pragma unroll
        for (int v = 0; v < 8; ++v) {
            int m = m0 + v + 8 * hi;      // m = b*T + t
            int b = m >> 9;               // / T
            int t = m & (TT - 1);         // % T
            xh[((size_t)t * BB + b) * HH + col] = acc[nt][v] + bias;   // [T,B,H]
        }
    }
}

// ---------------------------------------------------------------------------
// Phase 2 (one timestep): h_new = tanh(xh_t + h @ Wh^T)   (M=B=64, K=N=H)
// Block = 8 waves on one 16x64 tile, K split 8 ways (128 each);
// partials reduced through 32KB LDS. 64 blocks -> 512 waves.
// ---------------------------------------------------------------------------
__global__ __launch_bounds__(256, 1) __attribute__((amdgpu_waves_per_eu(1)))
void rnn_step_kernel(const _Float16* __restrict__ hsrc,
                     const _Float16* __restrict__ Whf,
                     const float* __restrict__ xh_t,
                     _Float16* __restrict__ hdst,
                     _Float16* __restrict__ hs_t,
                     float* __restrict__ hf32) {
    __shared__ float red[8 * 16 * 64];
    int lane = threadIdx.x & 31;
    int wave = threadIdx.x >> 5;            // K-chunk owner
    int m0 = (blockIdx.x & 3) * 16;         // B=64 -> 4 row tiles
    int n0 = (blockIdx.x >> 2) * 64;        // H=1024 -> 16 col strips
    int kbase = wave * 128;

    v8f acc[4] = {};
    gemm_strip<128>(hsrc + kbase, HH, m0, Whf + kbase, HH, n0, lane, acc);

    // dump partial 16x64 tile to LDS
    int hi = lane >> 4, c = lane & 15;
    float* my = red + wave * (16 * 64);
#pragma unroll
    for (int nt = 0; nt < 4; ++nt)
#pragma unroll
        for (int v = 0; v < 8; ++v)
            my[(v + 8 * hi) * 64 + nt * 16 + c] = acc[nt][v];
    __syncthreads();

    // reduce 8 partials, add xh_t, tanh, fan out
    for (int e = threadIdx.x; e < 16 * 64; e += 256) {
        float s = 0.f;
#pragma unroll
        for (int w = 0; w < 8; ++w) s += red[w * (16 * 64) + e];
        int row = e >> 6, col = e & 63;
        size_t idx = (size_t)(m0 + row) * HH + (n0 + col);
        float val = tanhf(s + xh_t[idx]);
        _Float16 hv = (_Float16)val;
        hdst[idx] = hv;     // next step's A operand
        hs_t[idx] = hv;     // saved for output projection
        hf32[idx] = val;    // f32 copy (final one = h_last)
    }
}

// ---------------------------------------------------------------------------
// Phase 3: y[b,t,:] = hs[t,b,:] @ Wy^T + by    (M=T*B, K=H, N=O)
// grid: (O/64, M/128), block 256.
// ---------------------------------------------------------------------------
__global__ __launch_bounds__(256, 1) __attribute__((amdgpu_waves_per_eu(1)))
void y_gemm_kernel(const _Float16* __restrict__ hsf,
                   const _Float16* __restrict__ Wyf,
                   const float* __restrict__ by,
                   float* __restrict__ y) {
    int lane = threadIdx.x & 31;
    int wave = threadIdx.x >> 5;
    int m0 = (blockIdx.y * 8 + wave) * 16;   // m = t*B + b
    int n0 = blockIdx.x * 64;

    v8f acc[4] = {};
    gemm_strip<HH>(hsf, HH, m0, Wyf, HH, n0, lane, acc);

    int hi = lane >> 4, c = lane & 15;
#pragma unroll
    for (int nt = 0; nt < 4; ++nt) {
        int col = n0 + nt * 16 + c;
        float bias = by[col];
#pragma unroll
        for (int v = 0; v < 8; ++v) {
            int m = m0 + v + 8 * hi;
            int t = m >> 6;        // / B
            int b = m & (BB - 1);  // % B
            y[((size_t)b * TT + t) * OO + col] = acc[nt][v] + bias;  // [B,T,O]
        }
    }
}

// ---------------------------------------------------------------------------
extern "C" void kernel_launch(void* const* d_in, const int* in_sizes, int n_in,
                              void* d_out, int out_size, void* d_ws, size_t ws_size,
                              hipStream_t stream) {
    const float* x  = (const float*)d_in[0];
    const float* h0 = (const float*)d_in[1];
    const float* Wx = (const float*)d_in[2];
    const float* bx = (const float*)d_in[3];
    const float* Wh = (const float*)d_in[4];
    const float* bh = (const float*)d_in[5];
    const float* Wy = (const float*)d_in[6];
    const float* by = (const float*)d_in[7];
    float* out = (float*)d_out;

    char* ws = (char*)d_ws;
    size_t off = 0;
    auto alloc = [&](size_t bytes) -> void* {
        off = (off + 255) & ~(size_t)255;
        void* p = ws + off;
        off += bytes;
        return p;
    };
    _Float16* xf    = (_Float16*)alloc((size_t)BB * TT * II * sizeof(_Float16));
    _Float16* Wxf   = (_Float16*)alloc((size_t)HH * II * sizeof(_Float16));
    _Float16* Whf   = (_Float16*)alloc((size_t)HH * HH * sizeof(_Float16));
    _Float16* Wyf   = (_Float16*)alloc((size_t)OO * HH * sizeof(_Float16));
    float*    xh    = (float*)   alloc((size_t)TT * BB * HH * sizeof(float));
    _Float16* hsf   = (_Float16*)alloc((size_t)TT * BB * HH * sizeof(_Float16));
    _Float16* hbuf0 = (_Float16*)alloc((size_t)BB * HH * sizeof(_Float16));
    _Float16* hbuf1 = (_Float16*)alloc((size_t)BB * HH * sizeof(_Float16));
    float*    hf32  = (float*)   alloc((size_t)BB * HH * sizeof(float));
    _Float16* hping[2] = { hbuf0, hbuf1 };

    // --- convert fp32 operands to f16 workspace copies ---
    f32_to_f16_kernel<<<4096, 256, 0, stream>>>(x,  xf,  BB * TT * II);
    f32_to_f16_kernel<<<512,  256, 0, stream>>>(Wx, Wxf, HH * II);
    f32_to_f16_kernel<<<1024, 256, 0, stream>>>(Wh, Whf, HH * HH);
    f32_to_f16_kernel<<<512,  256, 0, stream>>>(Wy, Wyf, OO * HH);
    f32_to_f16_kernel<<<64,   256, 0, stream>>>(h0, hping[0], BB * HH);

    // --- phase 1: input projection for all timesteps ---
    {
        dim3 grid(HH / 64, (BB * TT) / 128);
        xh_gemm_kernel<<<grid, 256, 0, stream>>>(xf, Wxf, bx, bh, xh);
    }

    // --- phase 2: sequential scan over T (ping-pong h) ---
    for (int t = 0; t < TT; ++t) {
        rnn_step_kernel<<<64, 256, 0, stream>>>(
            hping[t & 1], Whf,
            xh  + (size_t)t * BB * HH,
            hping[(t + 1) & 1],
            hsf + (size_t)t * BB * HH,
            hf32);
    }

    // --- phase 3: output projection ---
    {
        dim3 grid(OO / 64, (TT * BB) / 128);
        y_gemm_kernel<<<grid, 256, 0, stream>>>(hsf, Wyf, by, out);
    }

    // --- h_last -> tail of d_out ---
    hipMemcpyAsync(out + (size_t)BB * TT * OO, hf32,
                   (size_t)BB * HH * sizeof(float),
                   hipMemcpyDeviceToDevice, stream);
}
